// ContrastiveHead_34119220199639
// MI455X (gfx1250) — matrix operand order repaired and checked by reading, not compiled, not run on hardware
//
#include <hip/hip_runtime.h>
#include <hip/hip_bf16.h>

// ---------------- constants ----------------
#define BATCH 4096
#define DIMD  2048
#define DIMH  2048
#define DIME  128
#define N2    8192          // 2*BATCH
#define INV_T (1.0f / 0.07f)

typedef __attribute__((ext_vector_type(16))) __bf16 v16bf;
typedef __attribute__((ext_vector_type(8)))  float  v8f;
typedef __attribute__((ext_vector_type(4)))  float  f4;

#define WMMA_BF16(A, B, C) \
    __builtin_amdgcn_wmma_f32_16x16x32_bf16(false, (A), false, (B), (short)0, (C), false, false)

// ---------------- fragment loaders (wave32, 16x16x32 bf16) ----------------
// A matrix 16x32 (MxK), row-major source, ld = row stride in elements.
// lane&15 -> row M; lanes<16: K = k0+{0..7, 16..23}; lanes>=16: K = k0+{8..15, 24..31}
__device__ inline v16bf load_frag_a(const __bf16* __restrict__ base, int ld, int row, int k0) {
    const int lane = threadIdx.x & 31;
    const __bf16* p = base + (size_t)(row + (lane & 15)) * ld + (k0 + ((lane >> 4) << 3));
    union { v16bf v; f4 f[2]; } u;
    u.f[0] = *(const f4*)p;          // 8 bf16 = 16 B
    u.f[1] = *(const f4*)(p + 16);   // K + 16
    return u.v;
}

// B matrix 32x16 (KxN) gathered from a row-major [N][K] source (i.e. B[k][n] = src[n][k]).
// lane&15 -> col N; lanes<16: K = k0+{0..7, 16..23}; lanes>=16: K = k0+{8..15, 24..31}
__device__ inline v16bf load_frag_b(const __bf16* __restrict__ base, int ld, int col0, int k0) {
    const int lane = threadIdx.x & 31;
    const __bf16* p = base + (size_t)(col0 + (lane & 15)) * ld + (k0 + ((lane >> 4) << 3));
    union { v16bf v; f4 f[2]; } u;
    u.f[0] = *(const f4*)p;
    u.f[1] = *(const f4*)(p + 16);
    return u.v;
}

// ---------------- stage 0: fp32 -> bf16 stacked inputs ----------------
__global__ __launch_bounds__(256) void cvt_inputs_kernel(const float* __restrict__ in1,
                                                         const float* __restrict__ in2,
                                                         __bf16* __restrict__ X) {
    const size_t half = (size_t)BATCH * DIMD;
    size_t base = ((size_t)blockIdx.x * blockDim.x + threadIdx.x) * 4;
    const float* src = (base < half) ? (in1 + base) : (in2 + (base - half));
    f4 v = *(const f4*)src;
    X[base + 0] = (__bf16)v.x;
    X[base + 1] = (__bf16)v.y;
    X[base + 2] = (__bf16)v.z;
    X[base + 3] = (__bf16)v.w;
}

// ---------------- stage 1: transpose + convert W[K][N] -> WT[N][K] bf16 ----------------
__global__ __launch_bounds__(256) void transpose_cvt_kernel(const float* __restrict__ W,
                                                            __bf16* __restrict__ WT,
                                                            int K, int N) {
    __shared__ float tile[32][33];
    const int n0 = blockIdx.x * 32, k0 = blockIdx.y * 32;
    const int tx = threadIdx.x, ty = threadIdx.y;      // 32 x 8
    #pragma unroll
    for (int i = 0; i < 32; i += 8)
        tile[ty + i][tx] = W[(size_t)(k0 + ty + i) * N + (n0 + tx)];
    __syncthreads();
    #pragma unroll
    for (int i = 0; i < 32; i += 8)
        WT[(size_t)(n0 + ty + i) * K + (k0 + tx)] = (__bf16)tile[tx][ty + i];
}

// ---------------- stage 2: bf16 WMMA GEMM:  C = act(A[MxK] @ WT[NxK]^T + bias) ----------------
// block = 128 threads (4 waves, 2x2), block tile 128x128, wave tile 64x64 (4x4 wmma tiles).
// Per k-step: 16 b128 loads (8 frags, all live) feeding 16 independent WMMA chains.
template <bool RELU, bool OUTF32>
__global__ __launch_bounds__(128) void gemm_bf16_kernel(const __bf16* __restrict__ A,
                                                        const __bf16* __restrict__ Bt,
                                                        const float* __restrict__ bias,
                                                        void* __restrict__ Cout,
                                                        int M, int N, int K) {
    const int lane = threadIdx.x & 31;
    const int wave = threadIdx.x >> 5;
    const int waveM = blockIdx.y * 128 + (wave & 1) * 64;
    const int waveN = blockIdx.x * 128 + (wave >> 1) * 64;

    v8f acc[4][4];
    const v8f vzero = {0.f, 0.f, 0.f, 0.f, 0.f, 0.f, 0.f, 0.f};
    #pragma unroll
    for (int s = 0; s < 4; ++s)
        #pragma unroll
        for (int t = 0; t < 4; ++t) acc[s][t] = vzero;

    for (int k0 = 0; k0 < K; k0 += 32) {
        v16bf a[4], b[4];
        #pragma unroll
        for (int s = 0; s < 4; ++s) a[s] = load_frag_a(A, K, waveM + s * 16, k0);
        #pragma unroll
        for (int t = 0; t < 4; ++t) b[t] = load_frag_b(Bt, K, waveN + t * 16, k0);
        #pragma unroll
        for (int s = 0; s < 4; ++s)
            #pragma unroll
            for (int t = 0; t < 4; ++t)
                acc[s][t] = WMMA_BF16(a[s], b[t], acc[s][t]);
    }

    // C/D layout: VGPR v -> row v (+8 for hi half-wave), lane&15 -> col
    const int nlo = lane & 15;
    const int hi8 = (lane >> 4) << 3;
    float bz[4];
    #pragma unroll
    for (int t = 0; t < 4; ++t) bz[t] = bias[waveN + t * 16 + nlo];

    #pragma unroll
    for (int s = 0; s < 4; ++s) {
        #pragma unroll
        for (int v = 0; v < 8; ++v) {
            const size_t row = (size_t)(waveM + s * 16 + hi8 + v) * N;
            #pragma unroll
            for (int t = 0; t < 4; ++t) {
                float val = acc[s][t][v] + bz[t];
                if (RELU) val = fmaxf(val, 0.0f);
                const size_t idx = row + (waveN + t * 16 + nlo);
                if (OUTF32) ((float*)Cout)[idx] = val;
                else        ((__bf16*)Cout)[idx] = (__bf16)val;
            }
        }
    }
}

// ---------------- stage 3: row L2-normalize, f32 -> bf16 ----------------
__global__ __launch_bounds__(256) void normalize_kernel(const float* __restrict__ Ef,
                                                        __bf16* __restrict__ F) {
    const int lane = threadIdx.x & 31;
    const int row  = blockIdx.x * 8 + (threadIdx.x >> 5);   // one wave per row
    const float* src = Ef + (size_t)row * DIME + lane * 4;
    f4 v = *(const f4*)src;
    float ss = v.x * v.x + v.y * v.y + v.z * v.z + v.w * v.w;
    #pragma unroll
    for (int off = 16; off >= 1; off >>= 1) ss += __shfl_xor(ss, off, 32);
    const float rn = rsqrtf(ss);
    __bf16* dst = F + (size_t)row * DIME + lane * 4;
    dst[0] = (__bf16)(v.x * rn);
    dst[1] = (__bf16)(v.y * rn);
    dst[2] = (__bf16)(v.z * rn);
    dst[3] = (__bf16)(v.w * rn);
}

// ---------------- stage 4: fused sim = F F^T / T, logsumexp, pos/self via diag tiles ------
// One block per 32 rows (2 row-tiles, af[2][4] held in regs); 8 waves stream all 512
// column tiles. Hot loop is BRANCH-FREE: 8 b128 loads -> 8 WMMAs (two independent
// chains sharing B-frags) -> 16 exp-adds. f is L2-normalized => sim/T <= 1/T, so the
// logsumexp max is the constant 1/T:  lse = 1/T + log(sum exp((c-1)/T)).
// The self-similarity term is included in the loop and subtracted exactly afterwards
// (recomputed by an identical WMMA -> bit-identical value). pos comes from the diagonal
// of the partner tile, also computed once after the loop by 4 dedicated waves.
__global__ __launch_bounds__(256) void loss_kernel(const __bf16* __restrict__ F,
                                                   float* __restrict__ rowLoss) {
    const int lane = threadIdx.x & 31;
    const int wave = threadIdx.x >> 5;
    const int i0   = blockIdx.x * 32;

    __shared__ float ssum[8][32];
    __shared__ float spos[32];    // raw partner dot (diag of partner tile)
    __shared__ float sself[32];   // raw self dot (diag of self tile)

    v16bf af[2][4];
    #pragma unroll
    for (int rt = 0; rt < 2; ++rt)
        #pragma unroll
        for (int q = 0; q < 4; ++q) af[rt][q] = load_frag_a(F, DIME, i0 + rt * 16, q * 32);

    float rsum[2][8];
    #pragma unroll
    for (int rt = 0; rt < 2; ++rt)
        #pragma unroll
        for (int v = 0; v < 8; ++v) rsum[rt][v] = 0.0f;

    const int nlo = lane & 15;
    const int hi8 = (lane >> 4) << 3;
    const v8f vzero = {0.f, 0.f, 0.f, 0.f, 0.f, 0.f, 0.f, 0.f};

    // ---- hot loop: all 512 column tiles, self term included (corrected later) ----
    for (int jt = wave; jt < (N2 / 16); jt += 8) {
        const int jc = jt * 16;
        v16bf b0 = load_frag_b(F, DIME, jc, 0);
        v16bf b1 = load_frag_b(F, DIME, jc, 32);
        v16bf b2 = load_frag_b(F, DIME, jc, 64);
        v16bf b3 = load_frag_b(F, DIME, jc, 96);
        v8f a0 = vzero, a1 = vzero;            // two independent chains keep b0..b3 live
        a0 = WMMA_BF16(af[0][0], b0, a0);  a1 = WMMA_BF16(af[1][0], b0, a1);
        a0 = WMMA_BF16(af[0][1], b1, a0);  a1 = WMMA_BF16(af[1][1], b1, a1);
        a0 = WMMA_BF16(af[0][2], b2, a0);  a1 = WMMA_BF16(af[1][2], b2, a1);
        a0 = WMMA_BF16(af[0][3], b3, a0);  a1 = WMMA_BF16(af[1][3], b3, a1);
        #pragma unroll
        for (int v = 0; v < 8; ++v) {
            rsum[0][v] += __expf((a0[v] - 1.0f) * INV_T);
            rsum[1][v] += __expf((a1[v] - 1.0f) * INV_T);
        }
    }

    // ---- cross-lane sum within each 16-lane half (same row coverage) ----
    #pragma unroll
    for (int off = 1; off < 16; off <<= 1)
        #pragma unroll
        for (int rt = 0; rt < 2; ++rt)
            #pragma unroll
            for (int v = 0; v < 8; ++v) rsum[rt][v] += __shfl_xor(rsum[rt][v], off, 32);
    if (nlo == 0) {
        #pragma unroll
        for (int rt = 0; rt < 2; ++rt)
            #pragma unroll
            for (int v = 0; v < 8; ++v) ssum[wave][rt * 16 + hi8 + v] = rsum[rt][v];
    }

    // ---- diagonal tiles: waves 0,1 -> pos for row-tile 0,1; waves 2,3 -> self ----
    if (wave < 4) {
        const int rt    = wave & 1;
        const bool slf  = (wave >= 2);
        const int jc    = slf ? (i0 + rt * 16) : ((i0 + rt * 16 + BATCH) & (N2 - 1));
        v16bf b0 = load_frag_b(F, DIME, jc, 0);
        v16bf b1 = load_frag_b(F, DIME, jc, 32);
        v16bf b2 = load_frag_b(F, DIME, jc, 64);
        v16bf b3 = load_frag_b(F, DIME, jc, 96);
        v8f acc = vzero;
        acc = WMMA_BF16(af[rt][0], b0, acc);
        acc = WMMA_BF16(af[rt][1], b1, acc);
        acc = WMMA_BF16(af[rt][2], b2, acc);
        acc = WMMA_BF16(af[rt][3], b3, acc);
        // diagonal element (m == n) lives in lanes 0..7 (v=lane) and 24..31 (v=lane-24)
        float d = acc[0];
        #pragma unroll
        for (int v = 1; v < 8; ++v) d = ((nlo - hi8) == v) ? acc[v] : d;
        if ((lane < 8) || (lane >= 24)) (slf ? sself : spos)[rt * 16 + nlo] = d;
    }
    __syncthreads();

    // ---- finalize: subtract exact self term, add pos, logsumexp ----
    if (threadIdx.x < 32) {
        const int r = threadIdx.x;
        float s = 0.0f;
        #pragma unroll
        for (int w = 0; w < 8; ++w) s += ssum[w][r];
        s -= __expf((sself[r] - 1.0f) * INV_T);        // bit-exact cancellation
        rowLoss[i0 + r] = (INV_T + __logf(s)) - spos[r] * INV_T;
    }
}

// ---------------- stage 5: deterministic mean reduction ----------------
__global__ __launch_bounds__(256) void reduce_mean_kernel(const float* __restrict__ rl,
                                                          float* __restrict__ out) {
    __shared__ float sm[256];
    float a = 0.0f;
    for (int i = threadIdx.x; i < N2; i += 256) a += rl[i];
    sm[threadIdx.x] = a;
    __syncthreads();
    for (int s = 128; s > 0; s >>= 1) {
        if ((int)threadIdx.x < s) sm[threadIdx.x] += sm[threadIdx.x + s];
        __syncthreads();
    }
    if (threadIdx.x == 0) out[0] = sm[0] / (float)N2;
}

// ---------------- host launcher ----------------
extern "C" void kernel_launch(void* const* d_in, const int* in_sizes, int n_in,
                              void* d_out, int out_size, void* d_ws, size_t ws_size,
                              hipStream_t stream) {
    const float* in1 = (const float*)d_in[0];
    const float* in2 = (const float*)d_in[1];
    const float* W0  = (const float*)d_in[2];
    const float* b0  = (const float*)d_in[3];
    const float* W1  = (const float*)d_in[4];
    const float* b1  = (const float*)d_in[5];
    const float* W2  = (const float*)d_in[6];
    const float* b2  = (const float*)d_in[7];

    char* ws = (char*)d_ws;
    size_t off = 0;
    auto take = [&](size_t bytes) -> char* {
        char* p = ws + off;
        off += (bytes + 255) & ~(size_t)255;
        return p;
    };
    __bf16* Xbf = (__bf16*)take((size_t)N2 * DIMD * 2);   // reused as H2 after GEMM1 consumes it
    __bf16* W0T = (__bf16*)take((size_t)DIMH * DIMD * 2);
    __bf16* W1T = (__bf16*)take((size_t)DIMH * DIMH * 2);
    __bf16* W2T = (__bf16*)take((size_t)DIME * DIMH * 2);
    __bf16* H1  = (__bf16*)take((size_t)N2 * DIMH * 2);
    float*  Ef  = (float*) take((size_t)N2 * DIME * 4);
    __bf16* Fn  = (__bf16*)take((size_t)N2 * DIME * 2);
    float*  rL  = (float*) take((size_t)N2 * 4);
    __bf16* H2  = Xbf;   // alias: X no longer needed once H1 exists

    // stage 0: stack + convert inputs
    {
        const size_t nquads = (size_t)N2 * DIMD / 4;
        cvt_inputs_kernel<<<dim3((unsigned)(nquads / 256)), dim3(256), 0, stream>>>(in1, in2, Xbf);
    }
    // stage 1: weight transposes
    transpose_cvt_kernel<<<dim3(DIMH / 32, DIMD / 32), dim3(32, 8), 0, stream>>>(W0, W0T, DIMD, DIMH);
    transpose_cvt_kernel<<<dim3(DIMH / 32, DIMH / 32), dim3(32, 8), 0, stream>>>(W1, W1T, DIMH, DIMH);
    transpose_cvt_kernel<<<dim3(DIME / 32, DIMH / 32), dim3(32, 8), 0, stream>>>(W2, W2T, DIMH, DIME);

    // stage 2: MLP GEMMs (bf16 WMMA, fp32 accumulate), 128x128 block tiles, 128 threads
    gemm_bf16_kernel<true,  false><<<dim3(DIMH / 128, N2 / 128), dim3(128), 0, stream>>>(
        Xbf, W0T, b0, H1, N2, DIMH, DIMD);
    gemm_bf16_kernel<false, false><<<dim3(DIMH / 128, N2 / 128), dim3(128), 0, stream>>>(
        H1, W1T, b1, H2, N2, DIMH, DIMH);
    gemm_bf16_kernel<false, true><<<dim3(DIME / 128, N2 / 128), dim3(128), 0, stream>>>(
        H2, W2T, b2, Ef, N2, DIME, DIMH);

    // stage 3: normalize rows
    normalize_kernel<<<dim3(N2 / 8), dim3(256), 0, stream>>>(Ef, Fn);

    // stage 4: fused similarity + logsumexp + pos (32 rows per block)
    loss_kernel<<<dim3(N2 / 32), dim3(256), 0, stream>>>(Fn, rL);

    // stage 5: mean
    reduce_mean_kernel<<<dim3(1), dim3(256), 0, stream>>>(rL, (float*)d_out);
}
